// FrozenLSLModel_32512902431221
// MI455X (gfx1250) — matrix operand order repaired and checked by reading, not compiled, not run on hardware
//
#include <hip/hip_runtime.h>
#include <hip/hip_bf16.h>

// ---------------------------------------------------------------------------
// FrozenLSLModel collapsed to 64/4096-entry tables + WMMA GEMM + gather.
// ---------------------------------------------------------------------------

#define HID       768
#define MAX_BITS  64
#define NBATCH    32768
#define KT        (HID / 32)        // 24 K-tiles of 32 for f16 WMMA
#define NT1       (HID / 16)        // 48 N-tiles for GEMM1 (768 outputs)
#define NT2       (MAX_BITS / 16)   // 4  N-tiles for GEMM2 (64 logits)
#define LDS_STRIDE 776              // 768 + 8 halfs pad (byte stride 1552 -> bank spread)

typedef __attribute__((ext_vector_type(16))) _Float16 v16h;
typedef __attribute__((ext_vector_type(8)))  _Float16 v8h;
typedef __attribute__((ext_vector_type(8)))  float    v8f;

// ---------------------------------------------------------------------------
// Kernel 0: pack an f32 [K,N] (row-major, k-major) weight into f16 WMMA
// B-fragment order for V_WMMA_F32_16X16X32_F16.
// Tile (nt,kt) -> 512 halfs laid out as [lane][16]:
//   lanes 0-15 : N = nt*16+lane,    halfs j=0..15 = W[kt*32 + j     ][N]
//   lanes 16-31: N = nt*16+lane-16, halfs j=0..15 = W[kt*32 + 16 + j][N]
// Tile order: tile = nt*KT + kt  (K-tiles contiguous for streaming).
// ---------------------------------------------------------------------------
__global__ void pack_b_kernel(const float* __restrict__ W, _Float16* __restrict__ out,
                              int ktiles, int ntiles, int N) {
    int t = blockIdx.x * blockDim.x + threadIdx.x;
    int total = ktiles * ntiles * 32;
    if (t >= total) return;
    int lane = t & 31;
    int tile = t >> 5;
    int kt = tile % ktiles;
    int nt = tile / ktiles;
    int k0 = kt * 32 + ((lane & 16) ? 16 : 0);
    int n  = nt * 16 + (lane & 15);
    _Float16* dst = out + (size_t)tile * 512 + (size_t)lane * 16;
#pragma unroll
    for (int j = 0; j < 16; ++j)
        dst[j] = (_Float16)W[(size_t)(k0 + j) * N + n];
}

// ---------------------------------------------------------------------------
// Kernel 1: shift_decoder lookup table.  Only 64 distinct inputs exist
// (shift_amount in [0,64)), so run the 64->768->768->64 MLP once per value.
// 64 blocks x 256 threads, scalar f32 (38M MACs total - negligible).
// ---------------------------------------------------------------------------
__global__ __launch_bounds__(256)
void shift_decoder_kernel(const float* __restrict__ Ws0, const float* __restrict__ bs0,
                          const float* __restrict__ Ws1, const float* __restrict__ bs1,
                          const float* __restrict__ Ws2, const float* __restrict__ bs2,
                          int* __restrict__ pred) {
    __shared__ float h0[HID];
    __shared__ float h1[HID];
    __shared__ float logits[MAX_BITS];
    int s = blockIdx.x;     // shift value 0..63
    int t = threadIdx.x;

    // h0 = relu(bits(s) @ Ws0 + bs0): sum of rows of Ws0 for set bits (s < 64 -> bits 0..5)
    for (int j = t; j < HID; j += 256) {
        float acc = bs0[j];
#pragma unroll
        for (int k = 0; k < 6; ++k)
            if ((s >> k) & 1) acc += Ws0[k * HID + j];
        h0[j] = fmaxf(acc, 0.0f);
    }
    __syncthreads();
    // h1 = relu(h0 @ Ws1 + bs1)
    for (int j = t; j < HID; j += 256) {
        float acc = bs1[j];
        for (int k = 0; k < HID; ++k)
            acc = fmaf(h0[k], Ws1[(size_t)k * HID + j], acc);
        h1[j] = fmaxf(acc, 0.0f);
    }
    __syncthreads();
    // logits = h1 @ Ws2 + bs2
    if (t < MAX_BITS) {
        float acc = bs2[t];
        for (int k = 0; k < HID; ++k)
            acc = fmaf(h1[k], Ws2[(size_t)k * MAX_BITS + t], acc);
        logits[t] = acc;
    }
    __syncthreads();
    if (t == 0) {
        int best = 0;
        float bv = logits[0];
        for (int j = 1; j < MAX_BITS; ++j)
            if (logits[j] > bv) { bv = logits[j]; best = j; }   // first-max like jnp.argmax
        pred[s] = best;
    }
}

// ---------------------------------------------------------------------------
// A-fragment loader: V_WMMA_F32_16X16X32_F16 A operand in this code's contiguous k order (matches packed B):
//   lanes 0-15 : row M=lane,    halfs 0..15 = K 0..15
//   lanes 16-31: row M=lane-16, halfs 0..15 = K 16..31
// src points to row-major f16 LDS with LDS_STRIDE halfs per row.
// ---------------------------------------------------------------------------
__device__ inline v16h load_a_frag(const _Float16* __restrict__ src,
                                   int m0, int k0, int lane) {
    int row = m0 + (lane & 15);
    int kb  = (lane & 16) ? 16 : 0;
    const _Float16* ap = src + (size_t)row * LDS_STRIDE + k0 + kb;
    v8h lo = *(const v8h*)(ap);        // 16B-aligned (stride 1552B, k0 mult of 32, kb 0/16)
    v8h hi = *(const v8h*)(ap + 8);
    v16h a;
#pragma unroll
    for (int j = 0; j < 8; ++j) { a[j] = lo[j]; a[j + 8] = hi[j]; }
    return a;
}

// ---------------------------------------------------------------------------
// Kernel 2: index_net + validity_net tables over the 4096 distinct (p,i)
// combinations.  One block per p (64 blocks), 256 threads = 8 wave32.
//   h1[i,:] = relu(Wi0[i] + Wi0[64+p] + bi0)                (LDS f16, 64x768)
//   h2      = relu(h1 @ Wi1 + bi1)                          (WMMA, 64x768)
//   logits  = h2 @ Wi2 + bi2                                (WMMA, 64x64)
//   ptr[p*64+i]  = argmax(logits[i,:])
//   gate[p*64+i] = sigmoid(relu(Wv0[i]+Wv0[64+p]+bv0) @ Wv1 + bv1)
// Dynamic LDS: h1 + h2 (f16, padded) = 194 KB; logits alias h1 after GEMM1.
// ---------------------------------------------------------------------------
__global__ __launch_bounds__(256)
void index_table_kernel(const float* __restrict__ Wi0, const float* __restrict__ bi0,
                        const float* __restrict__ bi1,
                        const _Float16* __restrict__ Wi1p,   // packed 768x768
                        const _Float16* __restrict__ Wi2p,   // packed 768x64
                        const float* __restrict__ bi2,
                        const float* __restrict__ Wv0, const float* __restrict__ bv0,
                        const float* __restrict__ Wv1, const float* __restrict__ bv1,
                        int* __restrict__ ptr_out, float* __restrict__ gate_out) {
    extern __shared__ char smem[];
    _Float16* h1 = (_Float16*)smem;                 // 64 * LDS_STRIDE halfs
    _Float16* h2 = h1 + 64 * LDS_STRIDE;            // 64 * LDS_STRIDE halfs

    const int p    = blockIdx.x;                    // 0..63
    const int t    = threadIdx.x;
    const int wave = t >> 5;
    const int lane = t & 31;

    // ---- build h1 (coalesced f32 reads, f16 LDS writes) ----
    const float* w_shift = Wi0 + (size_t)(MAX_BITS + p) * HID;
    for (int i = wave; i < 64; i += 8) {
        const float* w_pos = Wi0 + (size_t)i * HID;
        for (int k = lane; k < HID; k += 32) {
            float v = w_pos[k] + w_shift[k] + bi0[k];
            h1[(size_t)i * LDS_STRIDE + k] = (_Float16)fmaxf(v, 0.0f);
        }
    }
    __syncthreads();

    // ---- GEMM1: h2 = relu(h1 @ Wi1 + bi1), 4 x 48 tiles of 16x16, K=768 ----
    for (int tile = wave; tile < 4 * NT1; tile += 8) {
        int tm = tile & 3;
        int tn = tile >> 2;
        v8f acc = {};
        const _Float16* bfrag = Wi1p + (size_t)(tn * KT) * 512 + (size_t)lane * 16;
        for (int kt = 0; kt < KT; ++kt) {
            v16h a = load_a_frag(h1, tm * 16, kt * 32, lane);
            v16h b = *(const v16h*)bfrag;           // 32B contiguous per lane
            bfrag += 512;
            acc = __builtin_amdgcn_wmma_f32_16x16x32_f16(
                false, a, false, b, (short)0, acc, false, false);
        }
        int n     = tn * 16 + (lane & 15);
        int mbase = tm * 16 + ((lane & 16) ? 8 : 0); // C/D layout: VGPR v -> M = mbase+v
        float bias = bi1[n];
#pragma unroll
        for (int v = 0; v < 8; ++v)
            h2[(size_t)(mbase + v) * LDS_STRIDE + n] = (_Float16)fmaxf(acc[v] + bias, 0.0f);
    }
    __syncthreads();            // h2 complete; h1 no longer read -> reuse as logits
    float* logits = (float*)h1; // 64*64 f32 = 16 KB inside old h1 space

    // ---- GEMM2: logits = h2 @ Wi2 + bi2, 4 x 4 tiles ----
    for (int tile = wave; tile < 4 * NT2; tile += 8) {
        int tm = tile & 3;
        int tn = tile >> 2;
        v8f acc = {};
        const _Float16* bfrag = Wi2p + (size_t)(tn * KT) * 512 + (size_t)lane * 16;
        for (int kt = 0; kt < KT; ++kt) {
            v16h a = load_a_frag(h2, tm * 16, kt * 32, lane);
            v16h b = *(const v16h*)bfrag;
            bfrag += 512;
            acc = __builtin_amdgcn_wmma_f32_16x16x32_f16(
                false, a, false, b, (short)0, acc, false, false);
        }
        int n     = tn * 16 + (lane & 15);
        int mbase = tm * 16 + ((lane & 16) ? 8 : 0);
        float bias = bi2[n];
#pragma unroll
        for (int v = 0; v < 8; ++v)
            logits[(mbase + v) * MAX_BITS + n] = acc[v] + bias;
    }
    __syncthreads();

    // ---- argmax per row -> ptr table (first-max semantics) ----
    if (t < 64) {
        const float* row = logits + t * MAX_BITS;
        int best = 0;
        float bv = row[0];
        for (int j = 1; j < MAX_BITS; ++j)
            if (row[j] > bv) { bv = row[j]; best = j; }
        ptr_out[p * 64 + t] = best;
    }

    // ---- validity net: 384 MACs/row, f32 scalar ----
    if (t < 64) {
        const int i = t;
        const float* wv_pos   = Wv0 + (size_t)i * (HID / 2);
        const float* wv_shift = Wv0 + (size_t)(MAX_BITS + p) * (HID / 2);
        float acc = 0.0f;
        for (int j = 0; j < HID / 2; ++j) {
            float hv = fmaxf(wv_pos[j] + wv_shift[j] + bv0[j], 0.0f);
            acc = fmaf(hv, Wv1[j], acc);
        }
        float valid = acc + bv1[0];
        gate_out[p * 64 + i] = 1.0f / (1.0f + expf(-valid));
    }
}

// ---------------------------------------------------------------------------
// Kernel 3: streaming output.  out[b,i] = a_bits[b, ptr[p,i]] * gate[p,i],
// p = pred[shift_amount[b]].  Coalesced writes; gathers stay within each
// row's 256B span (L0/L2 resident); tables are tiny and cache-resident.
// This is the only B=32768-scale kernel: ~16 MB traffic -> HBM-bound ~1us.
// ---------------------------------------------------------------------------
__global__ __launch_bounds__(256)
void gather_out_kernel(const float* __restrict__ a_bits,
                       const int* __restrict__ shift_amount,
                       const int* __restrict__ pred,
                       const int* __restrict__ ptr,
                       const float* __restrict__ gate,
                       float* __restrict__ out) {
    int t = blockIdx.x * blockDim.x + threadIdx.x;
    if (t >= NBATCH * MAX_BITS) return;
    int b = t >> 6;
    int i = t & 63;
    int p = pred[shift_amount[b] & 63];
    int r = p * 64 + i;
    out[t] = a_bits[(size_t)b * MAX_BITS + ptr[r]] * gate[r];
}

// ---------------------------------------------------------------------------
extern "C" void kernel_launch(void* const* d_in, const int* in_sizes, int n_in,
                              void* d_out, int out_size, void* d_ws, size_t ws_size,
                              hipStream_t stream) {
    const float* a_bits       = (const float*)d_in[0];
    const int*   shift_amount = (const int*)  d_in[1];
    const float* Ws0 = (const float*)d_in[2];
    const float* bs0 = (const float*)d_in[3];
    const float* Ws1 = (const float*)d_in[4];
    const float* bs1 = (const float*)d_in[5];
    const float* Ws2 = (const float*)d_in[6];
    const float* bs2 = (const float*)d_in[7];
    const float* Wi0 = (const float*)d_in[8];
    const float* bi0 = (const float*)d_in[9];
    const float* Wi1 = (const float*)d_in[10];
    const float* bi1 = (const float*)d_in[11];
    const float* Wi2 = (const float*)d_in[12];
    const float* bi2 = (const float*)d_in[13];
    const float* Wv0 = (const float*)d_in[14];
    const float* bv0 = (const float*)d_in[15];
    const float* Wv1 = (const float*)d_in[16];
    const float* bv1 = (const float*)d_in[17];
    float* out = (float*)d_out;

    // workspace layout
    char* ws = (char*)d_ws;
    size_t off = 0;
    _Float16* wi1p = (_Float16*)(ws + off); off += (size_t)HID * HID * sizeof(_Float16);      // 1.125 MB
    _Float16* wi2p = (_Float16*)(ws + off); off += (size_t)HID * MAX_BITS * sizeof(_Float16); // 96 KB
    int*   pred = (int*)  (ws + off); off += 64 * sizeof(int);
    int*   ptr  = (int*)  (ws + off); off += 64 * 64 * sizeof(int);
    float* gate = (float*)(ws + off); off += 64 * 64 * sizeof(float);
    (void)ws_size; (void)in_sizes; (void)n_in; (void)out_size;

    // 0) pack weights f32 -> f16 in WMMA B-fragment order
    {
        int tot1 = KT * NT1 * 32;   // 36864
        pack_b_kernel<<<(tot1 + 255) / 256, 256, 0, stream>>>(Wi1, wi1p, KT, NT1, HID);
        int tot2 = KT * NT2 * 32;   // 3072
        pack_b_kernel<<<(tot2 + 255) / 256, 256, 0, stream>>>(Wi2, wi2p, KT, NT2, MAX_BITS);
    }
    // 1) 64-entry shift-decoder table
    shift_decoder_kernel<<<64, 256, 0, stream>>>(Ws0, bs0, Ws1, bs1, Ws2, bs2, pred);
    // 2) 4096-entry (p,i) ptr/gate tables via WMMA
    {
        size_t smem = (size_t)2 * 64 * LDS_STRIDE * sizeof(_Float16);  // 194 KB
        index_table_kernel<<<64, 256, smem, stream>>>(Wi0, bi0, bi1, wi1p, wi2p, bi2,
                                                      Wv0, bv0, Wv1, bv1, ptr, gate);
    }
    // 3) streaming gather/scale for all 32768x64 outputs
    gather_out_kernel<<<(NBATCH * MAX_BITS) / 256, 256, 0, stream>>>(
        a_bits, shift_amount, pred, ptr, gate, out);
}
